// LstmClassifier_7249904796438
// MI455X (gfx1250) — matrix operand order repaired and checked by reading, not compiled, not run on hardware
//
#include <hip/hip_runtime.h>

#define NB       2048   // batch
#define IN_DIM_  1024
#define HID_     512
#define OUT_DIM_ 256
#define T_       64

typedef __attribute__((ext_vector_type(16))) _Float16     h16;
typedef __attribute__((ext_vector_type(8)))  _Float16     h8;
typedef __attribute__((ext_vector_type(8)))  float        f8;
typedef __attribute__((ext_vector_type(4))) unsigned int  u32x4;
typedef __attribute__((ext_vector_type(8)))  int          i32x8;
typedef __attribute__((ext_vector_type(4)))  int          i32x4;

#if defined(__has_builtin)
#if __has_builtin(__builtin_amdgcn_tensor_load_to_lds) && \
    __has_builtin(__builtin_amdgcn_s_wait_tensorcnt)
#define HAVE_TDM 1
#endif
#endif
#ifndef HAVE_TDM
#define HAVE_TDM 0
#endif

// Padded LDS layout: TDM pad_interval = 1024B, pad_amount = 16B.
// Half-index k within a K-long row lands at k + (k>>9)*8 halfs; row stride = LDSK(K).
__device__ __forceinline__ int kpos(int k) { return k + ((k >> 9) << 3); }
#define LDSK(K) ((K) + (((K) >> 9) << 3))

__device__ __forceinline__ h16 make_h16(h8 lo, h8 hi) {
  union { h16 v; h8 h[2]; } u; u.h[0] = lo; u.h[1] = hi; return u.v;
}

// A fragment (16x32 f16). lane<16: row=lane, e0..7=K0..7, e8..15=K16..23; lane>=16: K+8.
__device__ __forceinline__ h16 lds_a_frag(const _Float16* sA, int r0, int k0, int ldk, int lane) {
  const _Float16* p = sA + (r0 + (lane & 15)) * ldk + kpos(k0) + ((lane >> 4) << 3);
  return make_h16(*(const h8*)p, *(const h8*)(p + 16));
}
// B fragment (32x16 f16). lane L: col=L&15; e0..7=K0..7, e8..15=K16..23 (L<16); lane>=16: K+8.
__device__ __forceinline__ h16 lds_b_frag(const _Float16* sB, int c0, int k0, int ldk, int lane) {
  const _Float16* p = sB + (c0 + (lane & 15)) * ldk + kpos(k0) + ((lane >> 4) << 3);
  return make_h16(*(const h8*)p, *(const h8*)(p + 16));
}

__device__ __forceinline__ float sigmoid_(float x) { return 1.0f / (1.0f + __expf(-x)); }
__device__ __forceinline__ float tanh_(float x) {
  float e = __expf(-2.0f * fabsf(x));
  float r = (1.0f - e) / (1.0f + e);
  return copysignf(r, x);
}

#if HAVE_TDM
// Tensor DMA descriptor per CDNA5 ISA ch.8 (D# groups 0..3), 2-byte elements,
// tile = (dim0 K-halfs, dim1 rows, dim2 blocks[0=2D]), strides in elements.
// pad_enable, interval code 7 (1024B), amount code 3 (16B) -> matches kpos()/LDSK().
__device__ __forceinline__ void tdm_load(uint32_t lds_off, const _Float16* gptr,
                                         uint32_t dim0, uint32_t dim1, uint32_t dim2,
                                         uint64_t stride0, uint64_t stride1) {
  uint64_t ga = (uint64_t)(uintptr_t)gptr;
  u32x4 g0 = { 1u,                                   // count=1, user mode
               lds_off,
               (uint32_t)ga,
               (uint32_t)((ga >> 32) & 0x01ffffffu) | 0x80000000u };  // addr[56:32] | type=2
  i32x8 g1 = { (int)((1u << 16) | (1u << 20) | (7u << 22) | (3u << 25)),
               (int)((dim0 & 0xffffu) << 16),                               // tensor_dim0 lo
               (int)(((dim0 >> 16) & 0xffffu) | ((dim1 & 0xffffu) << 16)),  // dim0 hi | tensor_dim1 lo
               (int)(((dim1 >> 16) & 0xffffu) | (dim0 << 16)),              // dim1 hi | tile_dim0
               (int)(dim1 | (dim2 << 16)),                                  // tile_dim1 | tile_dim2
               (int)(stride0 & 0xffffffffu),
               (int)(((stride0 >> 32) & 0xffffu) | ((stride1 & 0xffffu) << 16)),
               (int)((stride1 >> 16) & 0xffffffffu) };
  i32x4 g2 = { (int)dim2, 0, 0, 0 };
  i32x4 g3 = { 0, 0, 0, 0 };
#if __clang_major__ >= 23
  i32x8 gz = { 0, 0, 0, 0, 0, 0, 0, 0 };
  __builtin_amdgcn_tensor_load_to_lds(g0, g1, g2, g3, gz, 0);
#else
  __builtin_amdgcn_tensor_load_to_lds(g0, g1, g2, g3, 0);
#endif
}
#endif

// ---------------- prep: f32 -> f16 conversions, weight/bias folding ----------------
__global__ __launch_bounds__(256) void prep_kernel(
    const float* __restrict__ x, const float* __restrict__ Win,
    const float* __restrict__ Wih, const float* __restrict__ Whh,
    const float* __restrict__ bih, const float* __restrict__ bhh,
    const float* __restrict__ Wout,
    _Float16* __restrict__ xb, _Float16* __restrict__ winb,
    _Float16* __restrict__ wcb, float* __restrict__ bcb,
    _Float16* __restrict__ woutb) {
  size_t id = (size_t)blockIdx.x * 256 + threadIdx.x;
  if (id < (size_t)NB * IN_DIM_)    xb[id]    = (_Float16)x[id];
  if (id < (size_t)HID_ * IN_DIM_)  winb[id]  = (_Float16)Win[id];
  if (id < (size_t)4 * HID_ * HID_) wcb[id]   = (_Float16)(Wih[id] + Whh[id]);
  if (id < (size_t)4 * HID_)        bcb[id]   = bih[id] + bhh[id];
  if (id < (size_t)OUT_DIM_ * HID_) woutb[id] = (_Float16)Wout[id];
}

// Cooperative fallback loader: rows*K halfs, 16B chunks, padded layout.
__device__ __forceinline__ void coop_load(_Float16* dst, const _Float16* src,
                                          int rows, int K, int lds_ldk, size_t src_ld,
                                          int tid) {
  const int cpr = K >> 3;  // 8-half chunks per row
  for (int c = tid; c < rows * cpr; c += 256) {
    int row = c / cpr, off = (c % cpr) << 3;
    *(h8*)(dst + row * lds_ldk + kpos(off)) = *(const h8*)(src + (size_t)row * src_ld + off);
  }
}

// ---------------- input projection: h0 = relu(x @ Win^T + b_in), c = 0 ----------------
__global__ __launch_bounds__(256) void inproj_kernel(
    const _Float16* __restrict__ X,     // [NB][IN_DIM]
    const _Float16* __restrict__ Win,   // [HID][IN_DIM]
    const float* __restrict__ bin,
    _Float16* __restrict__ H0,          // [NB][HID]
    float* __restrict__ C) {            // [NB][HID]
  extern __shared__ __align__(16) char smem_raw[];
  _Float16* sA = (_Float16*)smem_raw;                 // 64 x LDSK(1024)
  _Float16* sB = sA + 64 * LDSK(IN_DIM_);             // 32 x LDSK(1024)
  const int tid = threadIdx.x, lane = tid & 31, w = tid >> 5;
  const int mw = w & 3, jw = w >> 2;
  const int m0wg = blockIdx.x * 64, n0wg = blockIdx.y * 32;

#if HAVE_TDM
  if (w == 0) {
    uint32_t base = (uint32_t)(uintptr_t)sA;
    tdm_load(base, X + (size_t)m0wg * IN_DIM_, IN_DIM_, 64, 0, IN_DIM_, 0);
    tdm_load(base + (uint32_t)(64 * LDSK(IN_DIM_) * 2), Win + (size_t)n0wg * IN_DIM_,
             IN_DIM_, 32, 0, IN_DIM_, 0);
    __builtin_amdgcn_s_wait_tensorcnt(0);
  }
#else
  coop_load(sA, X + (size_t)m0wg * IN_DIM_, 64, IN_DIM_, LDSK(IN_DIM_), IN_DIM_, tid);
  coop_load(sB, Win + (size_t)n0wg * IN_DIM_, 32, IN_DIM_, LDSK(IN_DIM_), IN_DIM_, tid);
#endif
  __syncthreads();

  f8 acc = {};
#pragma unroll 4
  for (int k0 = 0; k0 < IN_DIM_; k0 += 32) {
    h16 a = lds_a_frag(sA, mw * 16, k0, LDSK(IN_DIM_), lane);
    h16 b = lds_b_frag(sB, jw * 16, k0, LDSK(IN_DIM_), lane);
    acc = __builtin_amdgcn_wmma_f32_16x16x32_f16(false, a, false, b, (short)0, acc,
                                                 false, false);
  }
  const int col = n0wg + jw * 16 + (lane & 15);
  const float bias = bin[col];
  const int rbase = m0wg + mw * 16 + ((lane >> 4) << 3);
#pragma unroll
  for (int v = 0; v < 8; ++v) {
    const int row = rbase + v;
    float h = acc[v] + bias;
    h = h > 0.0f ? h : 0.0f;
    H0[(size_t)row * HID_ + col] = (_Float16)h;
    C[(size_t)row * HID_ + col] = 0.0f;
  }
}

// ---------------- one LSTM step: gates = h @ (Wih+Whh)^T + bc; update c, h ----------------
__global__ __launch_bounds__(256) void lstm_step_kernel(
    const _Float16* __restrict__ Hin, int lda,  // [NB rows, stride lda]
    _Float16* __restrict__ Hout,                // row stride T_*HID_ (hs slice)
    float* __restrict__ C,                      // [NB][HID]
    const _Float16* __restrict__ Wc,            // [4*HID][HID]
    const float* __restrict__ bc) {             // [4*HID]
  extern __shared__ __align__(16) char smem_raw[];
  _Float16* sA = (_Float16*)smem_raw;           // 64  x LDSK(512)
  _Float16* sB = sA + 64 * LDSK(HID_);          // 128 x LDSK(512)  (4 gates x 32 cols)
  const int tid = threadIdx.x, lane = tid & 31, w = tid >> 5;
  const int mw = w & 3, jw = w >> 2;
  const int m0wg = blockIdx.x * 64, j0wg = blockIdx.y * 32;

#if HAVE_TDM
  if (w == 0) {
    uint32_t base = (uint32_t)(uintptr_t)sA;
    // A: 64 batch rows x full K=512
    tdm_load(base, Hin + (size_t)m0wg * lda, HID_, 64, 0, (uint64_t)lda, 0);
    // B: 3D tile — 32 weight rows per gate x 4 gate blocks, block stride 512*512
    tdm_load(base + (uint32_t)(64 * LDSK(HID_) * 2), Wc + (size_t)j0wg * HID_,
             HID_, 32, 4, HID_, (uint64_t)HID_ * HID_);
    __builtin_amdgcn_s_wait_tensorcnt(0);
  }
#else
  coop_load(sA, Hin + (size_t)m0wg * lda, 64, HID_, LDSK(HID_), (size_t)lda, tid);
  {  // 128 tile rows: gate = r>>5, within-gate col = r&31
    const int cpr = HID_ >> 3;
    for (int c = tid; c < 128 * cpr; c += 256) {
      int r = c / cpr, off = (c % cpr) << 3;
      int grow = (r >> 5) * HID_ + j0wg + (r & 31);
      *(h8*)(sB + r * LDSK(HID_) + kpos(off)) = *(const h8*)(Wc + (size_t)grow * HID_ + off);
    }
  }
#endif
  __syncthreads();

  f8 acc[4] = {};
#pragma unroll 4
  for (int k0 = 0; k0 < HID_; k0 += 32) {
    h16 a = lds_a_frag(sA, mw * 16, k0, LDSK(HID_), lane);
#pragma unroll
    for (int g = 0; g < 4; ++g) {
      h16 b = lds_b_frag(sB, g * 32 + jw * 16, k0, LDSK(HID_), lane);
      acc[g] = __builtin_amdgcn_wmma_f32_16x16x32_f16(false, a, false, b, (short)0,
                                                      acc[g], false, false);
    }
  }
  // epilogue in WMMA C layout: row = rbase+v, col = j0 + (lane&15)
  const int col = j0wg + jw * 16 + (lane & 15);
  const float bi = bc[col];
  const float bf = bc[HID_ + col];
  const float bg = bc[2 * HID_ + col];
  const float bo = bc[3 * HID_ + col];
  const int rbase = m0wg + mw * 16 + ((lane >> 4) << 3);
#pragma unroll
  for (int v = 0; v < 8; ++v) {
    const int row = rbase + v;
    float ig = sigmoid_(acc[0][v] + bi);
    float fg = sigmoid_(acc[1][v] + bf);
    float gg = tanh_(acc[2][v] + bg);
    float og = sigmoid_(acc[3][v] + bo);
    size_t cidx = (size_t)row * HID_ + col;
    float cv = fg * C[cidx] + ig * gg;
    C[cidx] = cv;
    Hout[(size_t)row * (T_ * HID_) + col] = (_Float16)(og * tanh_(cv));
  }
}

// ---------------- output projection: out = hs @ Wout^T + b_out ----------------
__global__ __launch_bounds__(256) void outproj_kernel(
    const _Float16* __restrict__ HS,    // [NB*T][HID] (flat view of hs)
    const _Float16* __restrict__ Wout,  // [OUT_DIM][HID]
    const float* __restrict__ bout,
    float* __restrict__ OUT) {          // [NB*T][OUT_DIM]
  extern __shared__ __align__(16) char smem_raw[];
  _Float16* sA = (_Float16*)smem_raw;           // 64 x LDSK(512)
  _Float16* sB = sA + 64 * LDSK(HID_);          // 32 x LDSK(512)
  const int tid = threadIdx.x, lane = tid & 31, w = tid >> 5;
  const int mw = w & 3, jw = w >> 2;
  const int m0wg = blockIdx.x * 64, n0wg = blockIdx.y * 32;

#if HAVE_TDM
  if (w == 0) {
    uint32_t base = (uint32_t)(uintptr_t)sA;
    tdm_load(base, HS + (size_t)m0wg * HID_, HID_, 64, 0, HID_, 0);
    tdm_load(base + (uint32_t)(64 * LDSK(HID_) * 2), Wout + (size_t)n0wg * HID_,
             HID_, 32, 0, HID_, 0);
    __builtin_amdgcn_s_wait_tensorcnt(0);
  }
#else
  coop_load(sA, HS + (size_t)m0wg * HID_, 64, HID_, LDSK(HID_), HID_, tid);
  coop_load(sB, Wout + (size_t)n0wg * HID_, 32, HID_, LDSK(HID_), HID_, tid);
#endif
  __syncthreads();

  f8 acc = {};
#pragma unroll 4
  for (int k0 = 0; k0 < HID_; k0 += 32) {
    h16 a = lds_a_frag(sA, mw * 16, k0, LDSK(HID_), lane);
    h16 b = lds_b_frag(sB, jw * 16, k0, LDSK(HID_), lane);
    acc = __builtin_amdgcn_wmma_f32_16x16x32_f16(false, a, false, b, (short)0, acc,
                                                 false, false);
  }
  const int col = n0wg + jw * 16 + (lane & 15);
  const float bias = bout[col];
  const int rbase = m0wg + mw * 16 + ((lane >> 4) << 3);
#pragma unroll
  for (int v = 0; v < 8; ++v) {
    const int row = rbase + v;
    OUT[(size_t)row * OUT_DIM_ + col] = acc[v] + bias;
  }
}

extern "C" void kernel_launch(void* const* d_in, const int* in_sizes, int n_in,
                              void* d_out, int out_size, void* d_ws, size_t ws_size,
                              hipStream_t stream) {
  const float* x     = (const float*)d_in[0];
  const float* W_in  = (const float*)d_in[1];
  const float* b_in  = (const float*)d_in[2];
  const float* W_ih  = (const float*)d_in[3];
  const float* b_ih  = (const float*)d_in[4];
  const float* W_hh  = (const float*)d_in[5];
  const float* b_hh  = (const float*)d_in[6];
  const float* W_out = (const float*)d_in[7];
  const float* b_out = (const float*)d_in[8];
  float* out = (float*)d_out;

  char* p = (char*)d_ws;
  auto carve = [&](size_t bytes) {
    char* r = p;
    p += (bytes + 255) & ~(size_t)255;
    return r;
  };
  _Float16* xb    = (_Float16*)carve((size_t)NB * IN_DIM_ * 2);
  _Float16* winb  = (_Float16*)carve((size_t)HID_ * IN_DIM_ * 2);
  _Float16* wcb   = (_Float16*)carve((size_t)4 * HID_ * HID_ * 2);
  float*    bcb   = (float*)carve((size_t)4 * HID_ * 4);
  _Float16* woutb = (_Float16*)carve((size_t)OUT_DIM_ * HID_ * 2);
  _Float16* h0b   = (_Float16*)carve((size_t)NB * HID_ * 2);
  float*    cbuf  = (float*)carve((size_t)NB * HID_ * 4);
  _Float16* hsb   = (_Float16*)carve((size_t)NB * T_ * HID_ * 2);
  (void)ws_size; (void)in_sizes; (void)n_in; (void)out_size;

  const size_t smem_in   = (size_t)(64 + 32) * LDSK(IN_DIM_) * 2;   // ~200 KB
  const size_t smem_step = (size_t)(64 + 128) * LDSK(HID_) * 2;     // ~200 KB
  const size_t smem_out  = (size_t)(64 + 32) * LDSK(HID_) * 2;      // ~100 KB

  prep_kernel<<<(NB * IN_DIM_) / 256, 256, 0, stream>>>(
      x, W_in, W_ih, W_hh, b_ih, b_hh, W_out, xb, winb, wcb, bcb, woutb);

  inproj_kernel<<<dim3(NB / 64, HID_ / 32), 256, smem_in, stream>>>(
      xb, winb, b_in, h0b, cbuf);

  for (int t = 0; t < T_; ++t) {
    const _Float16* Hin = (t == 0) ? h0b : (hsb + (size_t)(t - 1) * HID_);
    int lda = (t == 0) ? HID_ : (T_ * HID_);
    lstm_step_kernel<<<dim3(NB / 64, HID_ / 32), 256, smem_step, stream>>>(
        Hin, lda, hsb + (size_t)t * HID_, cbuf, wcb, bcb);
  }

  outproj_kernel<<<dim3((NB * T_) / 64, OUT_DIM_ / 32), 256, smem_out, stream>>>(
      hsb, woutb, b_out, out);
}
